// SelfAttention_33535104647232
// MI455X (gfx1250) — matrix operand order repaired and checked by reading, not compiled, not run on hardware
//
#include <hip/hip_runtime.h>
#include <hip/hip_bf16.h>
#include <cmath>

typedef __attribute__((ext_vector_type(16))) __bf16 v16bf;
typedef __attribute__((ext_vector_type(8)))  float  v8f;
typedef __attribute__((ext_vector_type(4)))  int    v4i;

union FragU { unsigned int u[8]; v16bf v; };
union U4    { uint4 q; __bf16 h[8]; };

#define B_  2
#define L_  2048
#define C_  1024
#define H_  16
#define HD_ 64

// ---------------- async global->LDS copy (gfx1250 ASYNCcnt path) ----------
#if defined(__has_builtin)
#  if __has_builtin(__builtin_amdgcn_global_load_async_to_lds_b128) && \
      __has_builtin(__builtin_amdgcn_s_wait_asynccnt)
#    define USE_ASYNC_LDS 1
#  endif
#endif
#ifndef USE_ASYNC_LDS
#  define USE_ASYNC_LDS 0
#endif

__device__ __forceinline__ void copy16_to_lds(__bf16* lds_dst, const __bf16* gsrc) {
#if USE_ASYNC_LDS
  __builtin_amdgcn_global_load_async_to_lds_b128(
      (v4i*)gsrc, (v4i*)lds_dst, 0, 0);
#else
  *(uint4*)lds_dst = *(const uint4*)gsrc;
#endif
}
__device__ __forceinline__ void async_join() {
#if USE_ASYNC_LDS
  __builtin_amdgcn_s_wait_asynccnt(0);
#endif
}

__device__ __forceinline__ float red_max16(float v) {
#pragma unroll
  for (int m = 8; m >= 1; m >>= 1) v = fmaxf(v, __shfl_xor(v, m, 32));
  return v;
}
__device__ __forceinline__ float red_sum16(float v) {
#pragma unroll
  for (int m = 8; m >= 1; m >>= 1) v += __shfl_xor(v, m, 32);
  return v;
}

// ---------------------------------------------------------------- convert
__global__ void cvt_f32_bf16_kernel(const float* __restrict__ s,
                                    __bf16* __restrict__ d, int n) {
  int i = blockIdx.x * blockDim.x + threadIdx.x;
  if (i < n) d[i] = (__bf16)s[i];
}

// ---------------------------------------------------------------- QKV GEMM
// A: [4096,1024] bf16, W: [1024,3072] bf16, bias f32[3072]
// out: Q/K/V as [B,H,L,HD] bf16 (bias folded in)
__global__ __launch_bounds__(256) void qkv_gemm_kernel(
    const __bf16* __restrict__ A, const __bf16* __restrict__ W,
    const float* __restrict__ bias,
    __bf16* __restrict__ Qo, __bf16* __restrict__ Ko, __bf16* __restrict__ Vo) {
  const int N = 3 * C_;
  const int K = C_;
  __shared__ __bf16 As[128 * 32];
  __shared__ __bf16 Bt[128 * 32];   // Bt[n][k]

  const int t = threadIdx.x;
  const int lane = t & 31, w = t >> 5;
  const int lane16 = lane & 15, half = lane >> 4;
  const int wm = w & 1, wn = w >> 1;
  const int m0 = blockIdx.x * 128, n0 = blockIdx.y * 128;

  const v8f vzero = {0.f, 0.f, 0.f, 0.f, 0.f, 0.f, 0.f, 0.f};
  v8f acc[4][2];
#pragma unroll
  for (int i = 0; i < 4; ++i)
#pragma unroll
    for (int j = 0; j < 2; ++j) acc[i][j] = vzero;

  const unsigned int* As_u = (const unsigned int*)As;
  const unsigned int* Bt_u = (const unsigned int*)Bt;

  for (int kt = 0; kt < K; kt += 32) {
    __syncthreads();
#pragma unroll
    for (int j = 0; j < 2; ++j) {               // A tile 128x32: async b128
      int idx = t + 256 * j;
      int row = idx >> 2, seg = idx & 3;
      copy16_to_lds(As + row * 32 + seg * 8,
                    A + (size_t)(m0 + row) * K + kt + seg * 8);
    }
#pragma unroll
    for (int j = 0; j < 2; ++j) {               // B tile 32x128 -> Bt[128][32]
      int idx = t + 256 * j;
      int k = idx >> 4, seg = idx & 15;
      U4 u;
      u.q = *(const uint4*)(W + (size_t)(kt + k) * N + n0 + seg * 8);
#pragma unroll
      for (int e = 0; e < 8; ++e) Bt[(seg * 8 + e) * 32 + k] = u.h[e];
    }
    if (kt + 32 < K) {                          // prefetch next tiles into L2
      __builtin_prefetch(A + (size_t)(m0 + (t >> 1)) * K + kt + 32, 0, 1);
      __builtin_prefetch(W + (size_t)(kt + 32 + (t >> 4)) * N + n0 + (t & 15) * 8, 0, 1);
    }
    async_join();
    __syncthreads();

    FragU a[4], b[2];
#pragma unroll
    for (int fm = 0; fm < 4; ++fm) {            // A frag: K = half*16 + 0..15
      int r = wm * 64 + fm * 16 + lane16;
#pragma unroll
      for (int p = 0; p < 4; ++p) {
        a[fm].u[p]     = As_u[r * 16 + half * 8 + p];
        a[fm].u[4 + p] = As_u[r * 16 + half * 8 + 4 + p];
      }
    }
#pragma unroll
    for (int fn = 0; fn < 2; ++fn) {            // B frag: K = half*16 + 0..15
      int n = wn * 32 + fn * 16 + lane16;
#pragma unroll
      for (int p = 0; p < 8; ++p) b[fn].u[p] = Bt_u[n * 16 + half * 8 + p];
    }
#pragma unroll
    for (int fm = 0; fm < 4; ++fm)
#pragma unroll
      for (int fn = 0; fn < 2; ++fn)
        acc[fm][fn] = __builtin_amdgcn_wmma_f32_16x16x32_bf16(
            false, a[fm].v, false, b[fn].v, (short)0, acc[fm][fn], false, false);
  }

#pragma unroll
  for (int fm = 0; fm < 4; ++fm) {
#pragma unroll
    for (int fn = 0; fn < 2; ++fn) {
      int gn = n0 + wn * 32 + fn * 16 + lane16;
      float bv = bias[gn];
      int s  = gn >> 10;
      int hh = (gn >> 6) & (H_ - 1);
      int d  = gn & (HD_ - 1);
      __bf16* dst = (s == 0) ? Qo : (s == 1) ? Ko : Vo;
#pragma unroll
      for (int r = 0; r < 8; ++r) {
        int gm = m0 + wm * 64 + fm * 16 + r + 8 * half;
        int bb = gm >> 11;
        int ll = gm & (L_ - 1);
        float val = acc[fm][fn][r] + bv;
        dst[(((size_t)(bb * H_ + hh)) * L_ + ll) * HD_ + d] = (__bf16)val;
      }
    }
  }
}

// ---------------------------------------------------------------- attention
// Q/K/V: [B*H, L, HD] bf16.  att: [B*L, C] bf16 (head-interleaved columns).
__global__ __launch_bounds__(256) void attn_kernel(
    const __bf16* __restrict__ Q, const __bf16* __restrict__ Km,
    const __bf16* __restrict__ Vm, const int* __restrict__ xmask,
    __bf16* __restrict__ att) {
  __shared__ __bf16 Kt[32 * 64];        // Kt[key][d]
  __shared__ __bf16 Vt[64 * 32];        // Vt[d][key]
  __shared__ __bf16 Ps[8 * 16 * 32];    // per-wave P scratch

  const int t = threadIdx.x;
  const int lane = t & 31, w = t >> 5;
  const int lane16 = lane & 15, half = lane >> 4;
  const int bh = blockIdx.x >> 4;
  const int chunk = blockIdx.x & 15;
  const int b = bh >> 4, h = bh & 15;
  const int i0 = chunk * 128 + w * 16;
  const size_t headoff = (size_t)bh * L_ * HD_;

  // q fragments (A layout), loaded directly from global
  const unsigned int* Qu = (const unsigned int*)(Q + headoff);
  FragU qf[2];
#pragma unroll
  for (int kk = 0; kk < 2; ++kk) {
    int row = i0 + lane16;
#pragma unroll
    for (int p = 0; p < 4; ++p) {
      qf[kk].u[p]     = Qu[row * 32 + kk * 16 + half * 8 + p];
      qf[kk].u[4 + p] = Qu[row * 32 + kk * 16 + half * 8 + 4 + p];
    }
  }

  const v8f vzero = {0.f, 0.f, 0.f, 0.f, 0.f, 0.f, 0.f, 0.f};
  v8f O[4];
#pragma unroll
  for (int f = 0; f < 4; ++f) O[f] = vzero;
  float mrow[8], lrow[8];
#pragma unroll
  for (int r = 0; r < 8; ++r) { mrow[r] = -3.0e38f; lrow[r] = 0.f; }

  const unsigned int* Kt_u = (const unsigned int*)Kt;
  const unsigned int* Vt_u = (const unsigned int*)Vt;
  __bf16* Pw = Ps + w * 512;
  const unsigned int* P_u = (const unsigned int*)Pw;
  const int* mk = xmask + b * L_;
  const float inv64 = 1.0f / 64.0f;   // q*HD^-0.5 then /sqrt(HD) quirk == /64

  for (int j0 = 0; j0 < L_; j0 += 32) {
    __syncthreads();                  // prior iteration done reading tiles
    {                                 // cooperative K/V tile loads
      int kk = t >> 3, seg = t & 7;
      copy16_to_lds(Kt + kk * 64 + seg * 8,                 // async path
                    Km + headoff + (size_t)(j0 + kk) * HD_ + seg * 8);
      U4 u;
      u.q = *(const uint4*)(Vm + headoff + (size_t)(j0 + kk) * HD_ + seg * 8);
#pragma unroll
      for (int e = 0; e < 8; ++e) Vt[(seg * 8 + e) * 32 + kk] = u.h[e];
      if (j0 + 32 < L_) {
        __builtin_prefetch(Km + headoff + (size_t)(j0 + 32 + kk) * HD_ + seg * 8, 0, 1);
        __builtin_prefetch(Vm + headoff + (size_t)(j0 + 32 + kk) * HD_ + seg * 8, 0, 1);
      }
    }
    async_join();
    __syncthreads();

    // S = q @ k^T  (two 16-key column blocks)
    v8f S[2];
#pragma unroll
    for (int nb = 0; nb < 2; ++nb) {
      S[nb] = vzero;
#pragma unroll
      for (int kk = 0; kk < 2; ++kk) {
        FragU bk;
        int kr = nb * 16 + lane16;    // B frag: column = key, K = HD dim
#pragma unroll
        for (int p = 0; p < 8; ++p)
          bk.u[p] = Kt_u[kr * 32 + kk * 16 + half * 8 + p];
        S[nb] = __builtin_amdgcn_wmma_f32_16x16x32_bf16(
            false, qf[kk].v, false, bk.v, (short)0, S[nb], false, false);
      }
    }

    // mask + scale + online softmax
    bool msk0 = (mk[j0 + lane16] == 1);
    bool msk1 = (mk[j0 + 16 + lane16] == 1);
    float alpha[8];
#pragma unroll
    for (int r = 0; r < 8; ++r) {
      float s0 = msk0 ? -1.25e9f : S[0][r] * inv64;
      float s1 = msk1 ? -1.25e9f : S[1][r] * inv64;
      float rm = red_max16(fmaxf(s0, s1));
      float mnew = fmaxf(mrow[r], rm);
      alpha[r] = __expf(mrow[r] - mnew);
      mrow[r] = mnew;
      float p0 = __expf(s0 - mnew);
      float p1 = __expf(s1 - mnew);
      lrow[r] = lrow[r] * alpha[r] + red_sum16(p0 + p1);
      int prow = r + 8 * half;
      Pw[prow * 32 + lane16]      = (__bf16)p0;
      Pw[prow * 32 + 16 + lane16] = (__bf16)p1;
    }
#pragma unroll
    for (int f = 0; f < 4; ++f)
#pragma unroll
      for (int r = 0; r < 8; ++r) O[f][r] *= alpha[r];

    __syncthreads();                  // P visible to whole wave

    FragU pf;                         // P as A fragment (16x32)
#pragma unroll
    for (int p = 0; p < 4; ++p) {
      pf.u[p]     = P_u[lane16 * 16 + half * 8 + p];
      pf.u[4 + p] = P_u[lane16 * 16 + half * 8 + 4 + p];
    }
#pragma unroll
    for (int f = 0; f < 4; ++f) {     // O += P @ V
      FragU bv;
      int d = f * 16 + lane16;
#pragma unroll
      for (int p = 0; p < 8; ++p) bv.u[p] = Vt_u[d * 16 + half * 8 + p];
      O[f] = __builtin_amdgcn_wmma_f32_16x16x32_bf16(
          false, pf.v, false, bv.v, (short)0, O[f], false, false);
    }
  }

  // normalize and write att [B*L, C] bf16
#pragma unroll
  for (int r = 0; r < 8; ++r) lrow[r] = 1.0f / lrow[r];
#pragma unroll
  for (int f = 0; f < 4; ++f) {
#pragma unroll
    for (int r = 0; r < 8; ++r) {
      int lq = i0 + r + 8 * half;
      int col = h * HD_ + f * 16 + lane16;
      att[(size_t)(b * L_ + lq) * C_ + col] = (__bf16)(O[f][r] * lrow[r]);
    }
  }
}

// ---------------------------------------------------------------- proj GEMM
__global__ __launch_bounds__(256) void proj_gemm_kernel(
    const __bf16* __restrict__ A, const __bf16* __restrict__ W,
    const float* __restrict__ bias, float* __restrict__ out) {
  const int N = C_, K = C_;
  __shared__ __bf16 As[128 * 32];
  __shared__ __bf16 Bt[128 * 32];

  const int t = threadIdx.x;
  const int lane = t & 31, w = t >> 5;
  const int lane16 = lane & 15, half = lane >> 4;
  const int wm = w & 1, wn = w >> 1;
  const int m0 = blockIdx.x * 128, n0 = blockIdx.y * 128;

  const v8f vzero = {0.f, 0.f, 0.f, 0.f, 0.f, 0.f, 0.f, 0.f};
  v8f acc[4][2];
#pragma unroll
  for (int i = 0; i < 4; ++i)
#pragma unroll
    for (int j = 0; j < 2; ++j) acc[i][j] = vzero;

  const unsigned int* As_u = (const unsigned int*)As;
  const unsigned int* Bt_u = (const unsigned int*)Bt;

  for (int kt = 0; kt < K; kt += 32) {
    __syncthreads();
#pragma unroll
    for (int j = 0; j < 2; ++j) {
      int idx = t + 256 * j;
      int row = idx >> 2, seg = idx & 3;
      copy16_to_lds(As + row * 32 + seg * 8,
                    A + (size_t)(m0 + row) * K + kt + seg * 8);
    }
#pragma unroll
    for (int j = 0; j < 2; ++j) {
      int idx = t + 256 * j;
      int k = idx >> 4, seg = idx & 15;
      U4 u;
      u.q = *(const uint4*)(W + (size_t)(kt + k) * N + n0 + seg * 8);
#pragma unroll
      for (int e = 0; e < 8; ++e) Bt[(seg * 8 + e) * 32 + k] = u.h[e];
    }
    if (kt + 32 < K) {
      __builtin_prefetch(A + (size_t)(m0 + (t >> 1)) * K + kt + 32, 0, 1);
      __builtin_prefetch(W + (size_t)(kt + 32 + (t >> 4)) * N + n0 + (t & 15) * 8, 0, 1);
    }
    async_join();
    __syncthreads();

    FragU a[4], b[2];
#pragma unroll
    for (int fm = 0; fm < 4; ++fm) {
      int r = wm * 64 + fm * 16 + lane16;
#pragma unroll
      for (int p = 0; p < 4; ++p) {
        a[fm].u[p]     = As_u[r * 16 + half * 8 + p];
        a[fm].u[4 + p] = As_u[r * 16 + half * 8 + 4 + p];
      }
    }
#pragma unroll
    for (int fn = 0; fn < 2; ++fn) {
      int n = wn * 32 + fn * 16 + lane16;
#pragma unroll
      for (int p = 0; p < 8; ++p) b[fn].u[p] = Bt_u[n * 16 + half * 8 + p];
    }
#pragma unroll
    for (int fm = 0; fm < 4; ++fm)
#pragma unroll
      for (int fn = 0; fn < 2; ++fn)
        acc[fm][fn] = __builtin_amdgcn_wmma_f32_16x16x32_bf16(
            false, a[fm].v, false, b[fn].v, (short)0, acc[fm][fn], false, false);
  }

#pragma unroll
  for (int fm = 0; fm < 4; ++fm) {
#pragma unroll
    for (int fn = 0; fn < 2; ++fn) {
      int gn = n0 + wn * 32 + fn * 16 + lane16;
      float bv = bias[gn];
#pragma unroll
      for (int r = 0; r < 8; ++r) {
        int gm = m0 + wm * 64 + fm * 16 + r + 8 * half;
        out[(size_t)gm * C_ + gn] = acc[fm][fn][r] + bv;
      }
    }
  }
}

// ---------------------------------------------------------------- launch
extern "C" void kernel_launch(void* const* d_in, const int* in_sizes, int n_in,
                              void* d_out, int out_size, void* d_ws, size_t ws_size,
                              hipStream_t stream) {
  (void)in_sizes; (void)n_in; (void)out_size; (void)ws_size;
  const float* input_x = (const float*)d_in[0];
  const int*   x_mask  = (const int*)d_in[1];
  const float* W_qkv   = (const float*)d_in[2];
  const float* b_qkv   = (const float*)d_in[3];
  const float* W_proj  = (const float*)d_in[4];
  const float* b_proj  = (const float*)d_in[5];
  float* out = (float*)d_out;

  char* ws = (char*)d_ws;
  __bf16* Xb  = (__bf16*)(ws);                        // 8 MiB (reused for att out)
  __bf16* Wqb = (__bf16*)(ws + (size_t)(8u  << 20));  // 6 MiB
  __bf16* Wpb = (__bf16*)(ws + (size_t)(14u << 20));  // 2 MiB
  __bf16* Qb  = (__bf16*)(ws + (size_t)(16u << 20));  // 8 MiB
  __bf16* Kb  = (__bf16*)(ws + (size_t)(24u << 20));  // 8 MiB
  __bf16* Vb  = (__bf16*)(ws + (size_t)(32u << 20));  // 8 MiB

  const int nX  = B_ * L_ * C_;        // 4194304
  const int nWq = C_ * 3 * C_;         // 3145728
  const int nWp = C_ * C_;             // 1048576

  cvt_f32_bf16_kernel<<<(nX  + 255) / 256, 256, 0, stream>>>(input_x, Xb, nX);
  cvt_f32_bf16_kernel<<<(nWq + 255) / 256, 256, 0, stream>>>(W_qkv, Wqb, nWq);
  cvt_f32_bf16_kernel<<<(nWp + 255) / 256, 256, 0, stream>>>(W_proj, Wpb, nWp);

  qkv_gemm_kernel<<<dim3(32, 24), 256, 0, stream>>>(Xb, Wqb, b_qkv, Qb, Kb, Vb);

  attn_kernel<<<dim3(B_ * H_ * (L_ / 128)), 256, 0, stream>>>(Qb, Kb, Vb, x_mask, Xb);

  proj_gemm_kernel<<<dim3(32, 8), 256, 0, stream>>>(Xb, Wpb, b_proj, out);
}
